// BiSDA_37160057045272
// MI455X (gfx1250) — matrix operand order repaired and checked by reading, not compiled, not run on hardware
//
#include <hip/hip_runtime.h>
#include <math.h>

// ---------------------------------------------------------------------------
// BiSDA pipeline for MI455X (gfx1250, wave32, WMMA).
// conv/cdc + channel GEMMs run on v_wmma_f32_16x16x32_bf16 (f32 accumulate),
// activations staged to LDS via global_load_async_to_lds_b128 (ASYNCcnt).
// ---------------------------------------------------------------------------

typedef __attribute__((ext_vector_type(16))) __bf16 v16bf;
typedef __attribute__((ext_vector_type(8)))  float  v8f;
typedef __attribute__((ext_vector_type(4)))  unsigned int u32x4;

union AF { v16bf v; u32x4 q[2]; };

#define C_CH 256
#define SPN  8192      // 8*32*32
#define NIMG 16        // T*B
#define THETA_C 0.7f

#define WMMA_BF16(a, b, c) \
  __builtin_amdgcn_wmma_f32_16x16x32_bf16(false, (a), false, (b), (short)0, (c), false, false)

// Issue one 16-byte async global->LDS copy (per-lane), tracked by ASYNCcnt.
__device__ __forceinline__ void async_b128_to_lds(const void* gptr, void* lds_generic) {
  unsigned ldsoff = (unsigned)(size_t)lds_generic;   // flat LDS addr[31:0] = LDS offset
  asm volatile("global_load_async_to_lds_b128 %0, %1, off"
               :: "v"(ldsoff), "v"(gptr) : "memory");
}
__device__ __forceinline__ void wait_async0() {
  asm volatile("s_wait_asynccnt 0" ::: "memory");
}

// ---------------------------------------------------------------------------
// Pack cdc_t weights: [O][I][3][3][3] f32 -> [tap][O][I] bf16, center tap
// corrected by -THETA * kdiff (kdiff = sum over kh,kw of kd=0 and kd=2 planes)
// ---------------------------------------------------------------------------
__global__ __launch_bounds__(256) void pack_qw_kernel(
    const float* __restrict__ w, __bf16* __restrict__ wp)
{
  int o = blockIdx.x;       // 256
  int i = threadIdx.x;      // 256
  const float* wb = w + ((long)(o * C_CH + i) * 27);
  float kdiff = 0.f;
  for (int kh = 0; kh < 3; ++kh)
    for (int kw = 0; kw < 3; ++kw)
      kdiff += wb[0 * 9 + kh * 3 + kw] + wb[2 * 9 + kh * 3 + kw];
  for (int tap = 0; tap < 27; ++tap) {
    float val = wb[tap];
    if (tap == 13) val -= THETA_C * kdiff;   // center (kd=1,kh=1,kw=1)
    wp[((long)tap * C_CH + o) * C_CH + i] = (__bf16)val;
  }
}

// [O][I] f32 -> [O][I] bf16 (for k_w / proj_w)
__global__ __launch_bounds__(256) void pack_mat_kernel(
    const float* __restrict__ w, __bf16* __restrict__ wp)
{
  int o = blockIdx.x, i = threadIdx.x;
  wp[o * C_CH + i] = (__bf16)w[o * C_CH + i];
}

// x [16][256][8192] f32 -> xb [16][8192][256] bf16 (channel-last)
__global__ __launch_bounds__(256) void cvt_kernel(
    const float* __restrict__ x, __bf16* __restrict__ xb)
{
  long idx = (long)blockIdx.x * 256 + threadIdx.x;   // dst linear order
  int c  = (int)(idx & 255);
  long r = idx >> 8;
  int sp = (int)(r & (SPN - 1));
  int n  = (int)(r >> 13);
  xb[idx] = (__bf16)x[((long)n * C_CH + c) * SPN + sp];
}

// ---------------------------------------------------------------------------
// Implicit-GEMM 3D conv (cdc folded into weights).
// Block = 16 output positions (16 consecutive w) x 256 out channels.
// Per valid (kd,kh): stage halo row [w0-1 .. w0+16] x 256ch (9216B) into LDS
// via async-to-LDS, then 3 kw taps x 8 K-steps x 2 WMMA per wave from LDS.
// ---------------------------------------------------------------------------
__global__ __launch_bounds__(256) void conv_wmma_kernel(
    const __bf16* __restrict__ xb,   // [16][8192][256] channel-last
    const __bf16* __restrict__ wp,   // [27][256 o][256 i]
    float* __restrict__ out)         // [16][256][8192]
{
  __shared__ u32x4 smemq[576];       // 18 pos x 32 chunks (8ch each)

  int bx = blockIdx.x;               // 8192 M-tiles
  int n  = bx >> 9;                  // 512 tiles per image
  int t  = bx & 511;
  int d  = t >> 6;
  int h  = (t >> 1) & 31;
  int w0 = (t & 1) << 4;

  int tid   = threadIdx.x;
  int wave  = tid >> 5;
  int lane  = tid & 31;
  int khalf = lane >> 4;
  int mrow  = lane & 15;
  int o0    = wave * 32;

  v8f acc0 = {};
  v8f acc1 = {};
  const u32x4 zq = {0u, 0u, 0u, 0u};

  for (int kd = 0; kd < 3; ++kd) {
    int dd = d + kd - 1;
    if (dd < 0 || dd >= 8) continue;
    for (int kh = 0; kh < 3; ++kh) {
      int hh = h + kh - 1;
      if (hh < 0 || hh >= 32) continue;

      // ---- stage activation row (with w halo) into LDS ----
      for (int chunk = tid; chunk < 576; chunk += 256) {
        int pos = chunk >> 5;            // 0..17  -> ww = w0 + pos - 1
        int cc  = (chunk & 31) << 3;     // channel base of this 8ch chunk
        int ww  = w0 + pos - 1;
        if (ww >= 0 && ww < 32) {
          const __bf16* gp =
              xb + (((long)n * SPN + ((dd * 32 + hh) * 32 + ww)) * C_CH + cc);
          async_b128_to_lds(gp, &smemq[chunk]);
        } else {
          smemq[chunk] = zq;             // zero-pad halo
        }
      }
      wait_async0();
      __syncthreads();

      // ---- consume: 3 kw taps from the staged row ----
      int tap_base = (kd * 3 + kh) * 3;
      for (int kw = 0; kw < 3; ++kw) {
        int tap = tap_base + kw;
        const __bf16* brow = wp + ((long)tap * C_CH * C_CH);
        if (tap < 26)
          __builtin_prefetch(wp + ((long)(tap + 1) * C_CH * C_CH) +
                             (long)(o0 + mrow) * C_CH);
        int pos = mrow + kw;             // staged-row index for output m=mrow
        for (int c0 = 0; c0 < C_CH; c0 += 32) {
          AF a;
          int cb = (c0 >> 3) + khalf;
          a.q[0] = smemq[pos * 32 + cb];       // K = c0 + khalf*8 .. +7
          a.q[1] = smemq[pos * 32 + cb + 2];   // K = c0 + 16 + khalf*8 .. +7
          AF b0, b1;
          const u32x4* pb0 =
              (const u32x4*)(brow + (long)(o0 + mrow) * C_CH + c0 + khalf * 8);
          b0.q[0] = pb0[0]; b0.q[1] = pb0[2];
          const u32x4* pb1 =
              (const u32x4*)(brow + (long)(o0 + 16 + mrow) * C_CH + c0 + khalf * 8);
          b1.q[0] = pb1[0]; b1.q[1] = pb1[2];

          acc0 = WMMA_BF16(a.v, b0.v, acc0);
          acc1 = WMMA_BF16(a.v, b1.v, acc1);
        }
      }
      __syncthreads();   // before next stage overwrites LDS
    }
  }

  int spb = (d * 32 + h) * 32 + w0;
  for (int r = 0; r < 8; ++r) {
    int m = r + khalf * 8;   // D layout: lane->N, vgpr r -> M (+8 for hi lanes)
    out[((long)n * C_CH + (o0 + mrow)) * SPN + spb + m]      = acc0[r];
    out[((long)n * C_CH + (o0 + 16 + mrow)) * SPN + spb + m] = acc1[r];
  }
}

// ---------------------------------------------------------------------------
// Channel GEMM: out[n][o][sp] = sum_i act[n][sp][i] * wgt[o][i]
// Activation tile (16 sp x 256 ch = 8KB) staged via async-to-LDS once.
// ---------------------------------------------------------------------------
__global__ __launch_bounds__(256) void gemm_wmma_kernel(
    const __bf16* __restrict__ act,  // [16][8192][256] channel-last
    const __bf16* __restrict__ wgt,  // [256 o][256 i] bf16
    float* __restrict__ out)         // [16][256][8192]
{
  __shared__ u32x4 smemq[512];       // 16 pos x 32 chunks

  int bx  = blockIdx.x;              // 8192 tiles
  int n   = bx >> 9;
  int sp0 = (bx & 511) << 4;

  int tid   = threadIdx.x;
  int wave  = tid >> 5;
  int lane  = tid & 31;
  int khalf = lane >> 4;
  int mrow  = lane & 15;
  int o0    = wave * 32;

  for (int chunk = tid; chunk < 512; chunk += 256) {
    int pos = chunk >> 5;
    int cc  = (chunk & 31) << 3;
    const __bf16* gp = act + ((long)n * SPN + sp0 + pos) * C_CH + cc;
    async_b128_to_lds(gp, &smemq[chunk]);
  }
  wait_async0();
  __syncthreads();

  v8f acc0 = {};
  v8f acc1 = {};
  for (int c0 = 0; c0 < C_CH; c0 += 32) {
    AF a;
    int cb = (c0 >> 3) + khalf;
    a.q[0] = smemq[mrow * 32 + cb];
    a.q[1] = smemq[mrow * 32 + cb + 2];
    AF b0, b1;
    const u32x4* pb0 = (const u32x4*)(wgt + (long)(o0 + mrow) * C_CH + c0 + khalf * 8);
    b0.q[0] = pb0[0]; b0.q[1] = pb0[2];
    const u32x4* pb1 = (const u32x4*)(wgt + (long)(o0 + 16 + mrow) * C_CH + c0 + khalf * 8);
    b1.q[0] = pb1[0]; b1.q[1] = pb1[2];
    acc0 = WMMA_BF16(a.v, b0.v, acc0);
    acc1 = WMMA_BF16(a.v, b1.v, acc1);
  }

  for (int r = 0; r < 8; ++r) {
    int m = r + khalf * 8;
    out[((long)n * C_CH + (o0 + mrow)) * SPN + sp0 + m]      = acc0[r];
    out[((long)n * C_CH + (o0 + 16 + mrow)) * SPN + sp0 + m] = acc1[r];
  }
}

// ---------------------------------------------------------------------------
// Per-channel BN statistics: mean + rsqrt(var + eps) over (n, sp).
// ---------------------------------------------------------------------------
__global__ __launch_bounds__(256) void stats_kernel(
    const float* __restrict__ x, float* __restrict__ mean, float* __restrict__ rstd)
{
  int c = blockIdx.x;   // 256
  float s = 0.f, s2 = 0.f;
  for (int idx = threadIdx.x; idx < NIMG * SPN; idx += 256) {
    int nn = idx >> 13;
    int sp = idx & (SPN - 1);
    float v = x[((long)nn * C_CH + c) * SPN + sp];
    s += v; s2 += v * v;
  }
  __shared__ float sh1[256], sh2[256];
  sh1[threadIdx.x] = s; sh2[threadIdx.x] = s2;
  __syncthreads();
  for (int off = 128; off > 0; off >>= 1) {
    if (threadIdx.x < off) {
      sh1[threadIdx.x] += sh1[threadIdx.x + off];
      sh2[threadIdx.x] += sh2[threadIdx.x + off];
    }
    __syncthreads();
  }
  if (threadIdx.x == 0) {
    float inv = 1.f / (float)(NIMG * SPN);
    float m   = sh1[0] * inv;
    float var = sh2[0] * inv - m * m;
    mean[c] = m;
    rstd[c] = rsqrtf(var + 1e-5f);
  }
}

// Region means (pre-BN): rs[b][win][c] = mean over T and window region.
__global__ __launch_bounds__(256) void region_kernel(
    const float* __restrict__ src, float* __restrict__ rs)
{
  int b   = blockIdx.x & 3;
  int win = blockIdx.x >> 2;          // 0..7 = (wt,wh,ww)
  int c   = threadIdx.x;
  int d0 = (win >> 2) * 4;
  int h0 = ((win >> 1) & 1) * 16;
  int w0 = (win & 1) * 16;
  float s = 0.f;
  for (int t = 0; t < 4; ++t)
    for (int dd = 0; dd < 4; ++dd)
      for (int hh = 0; hh < 16; ++hh) {
        const float* p = src + (((long)(t * 4 + b) * C_CH + c) * SPN) +
                         ((d0 + dd) * 32 + (h0 + hh)) * 32 + w0;
        for (int wwp = 0; wwp < 16; ++wwp) s += p[wwp];
      }
  rs[(b * 8 + win) * C_CH + c] = s * (1.f / 4096.f);
}

// Gain map: 8x8 region attention per batch, top-4 softmax -> per-window gain.
__global__ void gain_kernel(
    const float* __restrict__ rsq, const float* __restrict__ rsk,
    const float* __restrict__ qm, const float* __restrict__ qr,
    const float* __restrict__ qg, const float* __restrict__ qb,
    const float* __restrict__ km, const float* __restrict__ kr,
    const float* __restrict__ kg, const float* __restrict__ kb,
    float* __restrict__ gain)
{
  int b = threadIdx.x;
  if (b >= 4) return;
  float ar[8][8];
  for (int i = 0; i < 8; ++i)
    for (int j = 0; j < 8; ++j) {
      float s = 0.f;
      for (int c = 0; c < C_CH; ++c) {
        float qv = (rsq[(b * 8 + i) * C_CH + c] - qm[c]) * qr[c] * qg[c] + qb[c];
        float kv = (rsk[(b * 8 + j) * C_CH + c] - km[c]) * kr[c] * kg[c] + kb[c];
        s += qv * kv;
      }
      ar[i][j] = s * (1.f / 16.f);   // /sqrt(256)
    }
  float mask[8];
  for (int j = 0; j < 8; ++j) mask[j] = 0.f;
  for (int i = 0; i < 8; ++i) {
    float row[8];
    for (int j = 0; j < 8; ++j) row[j] = ar[i][j];
    int idx[4]; float val[4];
    for (int k2 = 0; k2 < 4; ++k2) {
      int bi = 0; float bv = row[0];
      for (int j = 1; j < 8; ++j)
        if (row[j] > bv) { bv = row[j]; bi = j; }
      idx[k2] = bi; val[k2] = bv; row[bi] = -3.4e38f;
    }
    float mx = val[0], e[4], se = 0.f;
    for (int k2 = 0; k2 < 4; ++k2) { e[k2] = expf(val[k2] - mx); se += e[k2]; }
    for (int k2 = 0; k2 < 4; ++k2) mask[idx[k2]] += e[k2] / se;
  }
  float ki[8], mx = 0.f;
  for (int j = 0; j < 8; ++j) { ki[j] = mask[j] * (1.f / 8.f); if (ki[j] > mx) mx = ki[j]; }
  for (int j = 0; j < 8; ++j)
    gain[b * 8 + j] = 0.5f + 2.f * (ki[j] / (mx + 1e-7f));
}

// ---------------------------------------------------------------------------
// Fused BN-affine (+optional window gain) + multi-step LIF over T.
// src [16][256][8192] f32 -> spikes dst [16][8192][256] bf16 (channel-last).
// ---------------------------------------------------------------------------
__global__ __launch_bounds__(256) void lif_kernel(
    const float* __restrict__ src,
    const float* __restrict__ mean, const float* __restrict__ rstd,
    const float* __restrict__ gamma, const float* __restrict__ beta,
    const float* __restrict__ gain,   // [4][8] or nullptr
    __bf16* __restrict__ dst)
{
  int idx = blockIdx.x * 256 + threadIdx.x;   // 4*256*8192
  int sp = idx & (SPN - 1);
  int c  = (idx >> 13) & 255;
  int b  = idx >> 21;
  float g = 1.f;
  if (gain) {
    int ds = sp >> 10, hr = (sp >> 5) & 31, wc = sp & 31;
    int win = ((ds >> 2) << 2) | ((hr >> 4) << 1) | (wc >> 4);
    g = gain[b * 8 + win];
  }
  float sc = gamma[c] * rstd[c];
  float of = beta[c] - mean[c] * sc;
  float vst = 0.f;
  for (int t = 0; t < 4; ++t) {
    float xv = (src[(((long)(t * 4 + b)) * C_CH + c) * SPN + sp] * sc + of) * g;
    vst += (xv - vst) * 0.5f;               // TAU = 2
    float s = (vst >= 1.0f) ? 1.f : 0.f;    // V_TH = 1
    vst *= (1.f - s);
    dst[(((long)(t * 4 + b)) * SPN + sp) * C_CH + c] = (__bf16)s;
  }
}

// ---------------------------------------------------------------------------
// Spike attention: attn = lif(sum_d q*k); out = lif(attn * v). Channel-last.
// One thread per (b, h, n).
// ---------------------------------------------------------------------------
__global__ __launch_bounds__(256) void attn_kernel(
    const __bf16* __restrict__ qs, const __bf16* __restrict__ ks,
    const __bf16* __restrict__ vs, __bf16* __restrict__ ob)
{
  int idx = blockIdx.x * 256 + threadIdx.x;   // 4*8*8192
  int n = idx & (SPN - 1);
  int h = (idx >> 13) & 7;
  int b = idx >> 16;

  float att_s[4];
  float vst = 0.f;
  for (int t = 0; t < 4; ++t) {
    long base = ((long)(t * 4 + b) * SPN + n) * C_CH + h * 32;
    float acc = 0.f;
    for (int dch = 0; dch < 32; ++dch)
      acc += (float)qs[base + dch] * (float)ks[base + dch];
    vst += (acc - vst) * 0.5f;
    float s = (vst >= 1.f) ? 1.f : 0.f;
    vst *= (1.f - s);
    att_s[t] = s;
  }
  for (int dch = 0; dch < 32; ++dch) {
    float v2 = 0.f;
    for (int t = 0; t < 4; ++t) {
      long base = ((long)(t * 4 + b) * SPN + n) * C_CH + h * 32 + dch;
      float o = att_s[t] * (float)vs[base];
      v2 += (o - v2) * 0.5f;
      float s = (v2 >= 1.f) ? 1.f : 0.f;
      v2 *= (1.f - s);
      ob[base] = (__bf16)s;
    }
  }
}

// Final BN apply in place on [16][256][8192].
__global__ __launch_bounds__(256) void bn_apply_kernel(
    float* __restrict__ y,
    const float* __restrict__ mean, const float* __restrict__ rstd,
    const float* __restrict__ gamma, const float* __restrict__ beta)
{
  long idx = (long)blockIdx.x * 256 + threadIdx.x;
  int c = (int)((idx >> 13) & 255);
  y[idx] = (y[idx] - mean[c]) * rstd[c] * gamma[c] + beta[c];
}

// ---------------------------------------------------------------------------
extern "C" void kernel_launch(void* const* d_in, const int* in_sizes, int n_in,
                              void* d_out, int out_size, void* d_ws, size_t ws_size,
                              hipStream_t stream)
{
  const float* x       = (const float*)d_in[0];
  const float* q_w     = (const float*)d_in[1];
  const float* k_w     = (const float*)d_in[2];
  const float* proj_w  = (const float*)d_in[3];
  const float* q_gamma = (const float*)d_in[4];
  const float* q_beta  = (const float*)d_in[5];
  const float* k_gamma = (const float*)d_in[6];
  const float* k_beta  = (const float*)d_in[7];
  const float* v_gamma = (const float*)d_in[8];
  const float* v_beta  = (const float*)d_in[9];
  const float* p_gamma = (const float*)d_in[10];
  const float* p_beta  = (const float*)d_in[11];

  char* ws = (char*)d_ws;
  const size_t SZ_BF  = (size_t)NIMG * SPN * C_CH * 2;   // 64 MiB
  const size_t SZ_F32 = (size_t)NIMG * SPN * C_CH * 4;   // 128 MiB

  __bf16* xb    = (__bf16*)(ws);
  float*  qc    = (float*)(ws + SZ_BF);
  float*  kl    = (float*)(ws + SZ_BF + SZ_F32);
  __bf16* wpack = (__bf16*)(ws + SZ_BF + 2 * SZ_F32);
  __bf16* kpack = (__bf16*)((char*)wpack + (size_t)27 * C_CH * C_CH * 2);
  __bf16* ppack = (__bf16*)((char*)kpack + (size_t)C_CH * C_CH * 2);
  float*  stats = (float*)((char*)ppack + (size_t)C_CH * C_CH * 2);
  float *qmean = stats,        *qrstd = stats + 256;
  float *kmean = stats + 512,  *krstd = stats + 768;
  float *vmean = stats + 1024, *vrstd = stats + 1280;
  float *pmean = stats + 1536, *prstd = stats + 1792;
  float* rsq  = stats + 2048;           // 4*8*256
  float* rsk  = rsq + 8192;
  float* gain = rsk + 8192;             // 4*8

  // Aliased buffers (lifetimes ordered by stream):
  __bf16* qs = (__bf16*)(ws);                         // over xb
  __bf16* ks = (__bf16*)(ws + SZ_BF);                 // over qc (1st half)
  __bf16* vs = (__bf16*)(ws + 2 * SZ_BF);             // over qc (2nd half)
  __bf16* ob = (__bf16*)(ws + SZ_BF + SZ_F32);        // over kl

  float* outp = (float*)d_out;

  // 1) weight packing + x conversion
  pack_qw_kernel<<<256, 256, 0, stream>>>(q_w, wpack);
  pack_mat_kernel<<<256, 256, 0, stream>>>(k_w, kpack);
  pack_mat_kernel<<<256, 256, 0, stream>>>(proj_w, ppack);
  cvt_kernel<<<131072, 256, 0, stream>>>(x, xb);

  // 2) heavy matrix work on WMMA
  conv_wmma_kernel<<<8192, 256, 0, stream>>>(xb, wpack, qc);
  gemm_wmma_kernel<<<8192, 256, 0, stream>>>(xb, kpack, kl);

  // 3) BN stats
  stats_kernel<<<256, 256, 0, stream>>>(qc, qmean, qrstd);
  stats_kernel<<<256, 256, 0, stream>>>(kl, kmean, krstd);
  stats_kernel<<<256, 256, 0, stream>>>(x,  vmean, vrstd);

  // 4) gain map
  region_kernel<<<32, 256, 0, stream>>>(qc, rsq);
  region_kernel<<<32, 256, 0, stream>>>(kl, rsk);
  gain_kernel<<<1, 32, 0, stream>>>(rsq, rsk, qmean, qrstd, q_gamma, q_beta,
                                    kmean, krstd, k_gamma, k_beta, gain);

  // 5) BN + LIF -> spikes (channel-last bf16)
  lif_kernel<<<32768, 256, 0, stream>>>(qc, qmean, qrstd, q_gamma, q_beta,
                                        (const float*)nullptr, qs);
  lif_kernel<<<32768, 256, 0, stream>>>(kl, kmean, krstd, k_gamma, k_beta,
                                        gain, ks);
  lif_kernel<<<32768, 256, 0, stream>>>(x, vmean, vrstd, v_gamma, v_beta,
                                        gain, vs);

  // 6) spike attention + output LIF
  attn_kernel<<<1024, 256, 0, stream>>>(qs, ks, vs, ob);

  // 7) projection GEMM (WMMA) -> d_out, then BN in place
  gemm_wmma_kernel<<<8192, 256, 0, stream>>>(ob, ppack, outp);
  stats_kernel<<<256, 256, 0, stream>>>(outp, pmean, prstd);
  bn_apply_kernel<<<131072, 256, 0, stream>>>(outp, pmean, prstd, p_gamma, p_beta);
}
